// CSA_79207786873391
// MI455X (gfx1250) — compile-verified, matrix operand order checked by reading, not run on hardware
//
#include <hip/hip_runtime.h>

#define DIM   384
#define HWSZ  16384
#define BATCH 8
#define HEADS 8
#define CH    48
#define MTOT  (BATCH * HWSZ)   // 131072 pixels

typedef __attribute__((ext_vector_type(16))) __bf16 v16bf;
typedef __attribute__((ext_vector_type(8)))  float  v8f;

// CDNA5 async global->LDS path (ASYNCcnt), with sync fallback.
#if __has_builtin(__builtin_amdgcn_global_load_async_to_lds_b128) && __has_builtin(__builtin_amdgcn_s_wait_asynccnt)
#define HAVE_ASYNC_LDS 1
typedef int v4i_vs __attribute__((vector_size(4 * sizeof(int))));
typedef __attribute__((address_space(1))) v4i_vs* as1_v4i;   // global
typedef __attribute__((address_space(3))) v4i_vs* as3_v4i;   // LDS
#define ASYNC_LDS_B128(gsrc, ldst) \
    __builtin_amdgcn_global_load_async_to_lds_b128((as1_v4i)(gsrc), (as3_v4i)(ldst), 0, 0)
#else
#define HAVE_ASYNC_LDS 0
#endif

// ---------- bf16 helpers (bit-level, no reliance on __bf16 arithmetic) ----------
__device__ __forceinline__ unsigned short f2bf(float f) {
    unsigned int u = __float_as_uint(f);
    unsigned int r = u + 0x7FFFu + ((u >> 16) & 1u);   // round-to-nearest-even
    return (unsigned short)(r >> 16);
}
__device__ __forceinline__ float bf2f(unsigned short h) {
    return __uint_as_float(((unsigned int)h) << 16);
}
__device__ __forceinline__ v8f vzero8() {
    v8f z = {0.f,0.f,0.f,0.f,0.f,0.f,0.f,0.f};
    return z;
}
__device__ __forceinline__ unsigned pack2(float a, float b) {
    return (unsigned)f2bf(a) | ((unsigned)f2bf(b) << 16);
}

// 16-bit matrix fragment load per CDNA5 ISA layout (A: 16x32, lane&15 = row,
// lanes 0-15 hold K 0-7 / 16-23, lanes 16-31 hold K 8-15 / 24-31).
// B fragment uses the mirrored layout (lane&15 = column, elements = K) so we
// stage B transposed ([n][k]) and use the same loader. Two 16B LDS/global loads.
__device__ __forceinline__ v16bf load_frag(const unsigned short* base, int rowStride, int lane) {
    int row = lane & 15;
    int kB  = (lane >> 4) << 3;
    const unsigned short* p = base + (size_t)row * rowStride + kB;
    union { uint4 u[2]; v16bf f; } u;
    u.u[0] = *(const uint4*)(p);
    u.u[1] = *(const uint4*)(p + 16);
    return u.f;
}

#define WMMA_BF16(A, B, Cacc) \
    __builtin_amdgcn_wmma_f32_16x16x32_bf16(false, (A), false, (B), (short)0, (Cacc), false, false)

// ---------------------------------------------------------------- zero init
__global__ void k_zero(float* p, int n) {
    int i = blockIdx.x * blockDim.x + threadIdx.x;
    if (i < n) p[i] = 0.f;
}

// ------------------------------------------------ LayerNorm (per pixel, C=384) -> bf16 [p][c]
__global__ void k_layernorm(const float* __restrict__ x, const float* __restrict__ g,
                            const float* __restrict__ b, unsigned short* __restrict__ act) {
    int tid  = threadIdx.x;
    int lane = tid & 31;
    int wave = tid >> 5;
    int p  = blockIdx.x * 8 + wave;          // one wave per pixel
    int bb = p >> 14;
    int hw = p & (HWSZ - 1);
    const float* base = x + (size_t)bb * DIM * HWSZ + hw;
    float v[12];
    float sum = 0.f, sq = 0.f;
#pragma unroll
    for (int i = 0; i < 12; i++) {
        float t = base[(size_t)(lane + 32 * i) * HWSZ];
        v[i] = t; sum += t; sq += t * t;
    }
#pragma unroll
    for (int off = 16; off > 0; off >>= 1) {
        sum += __shfl_xor(sum, off, 32);
        sq  += __shfl_xor(sq,  off, 32);
    }
    float mean = sum * (1.f / DIM);
    float var  = sq  * (1.f / DIM) - mean * mean;
    float rstd = rsqrtf(var + 1e-5f);
    unsigned short* dst = act + (size_t)p * DIM;
#pragma unroll
    for (int i = 0; i < 12; i++) {
        int c = lane + 32 * i;
        dst[c] = f2bf((v[i] - mean) * rstd * g[c] + b[c]);
    }
}

// --------------------------- 1x1-conv GEMM: out[p][n] = sum_k act[p][k]*W[n][k] (+bias)
// Block tile 128x64, 8 waves (4x2), each wave 32x32 = 2x2 WMMA tiles, K step 32.
// MODE 0: write bf16 NCHW planes (packed b128 stores).
// MODE 1: write f32 NCHW + bias + residual (float4 loads/stores).
template <int MODE>
__global__ void k_gemm384(const unsigned short* __restrict__ A, const float* __restrict__ W,
                          const float* __restrict__ bias, unsigned short* __restrict__ outBf,
                          float* __restrict__ outF, const float* __restrict__ resid) {
    __shared__ unsigned short As[128][40];   // pad 32->40 (80B rows, 16B aligned)
    __shared__ unsigned short Bs[64][40];    // stored transposed: [n][k]
    int tid  = threadIdx.x;
    int lane = tid & 31;
    int wave = tid >> 5;
    int wm = wave >> 1, wn = wave & 1;
    int m0 = blockIdx.x * 128;
    int n0 = blockIdx.y * 64;

    v8f acc[2][2];
    acc[0][0] = vzero8(); acc[0][1] = vzero8();
    acc[1][0] = vzero8(); acc[1][1] = vzero8();

    for (int kk = 0; kk < DIM; kk += 32) {
        __syncthreads();
        // stage A: 128 rows x 32 bf16, two 16B chunks per thread
#if HAVE_ASYNC_LDS
#pragma unroll
        for (int c = 0; c < 2; c++) {
            int chunk = tid * 2 + c;
            int r  = chunk >> 2;
            int cp = (chunk & 3) * 8;
            ASYNC_LDS_B128(A + (size_t)(m0 + r) * DIM + kk + cp, &As[r][cp]);
        }
#else
#pragma unroll
        for (int c = 0; c < 2; c++) {
            int chunk = tid * 2 + c;
            int r  = chunk >> 2;
            int cp = (chunk & 3) * 8;
            uint4 d = *(const uint4*)(A + (size_t)(m0 + r) * DIM + kk + cp);
            *(uint4*)(&As[r][cp]) = d;
        }
#endif
        // stage B: 64 rows x 32, convert f32 weights -> bf16 while staging
        {
            int r  = tid >> 2;
            int cp = (tid & 3) * 8;
            const float* wp = W + (size_t)(n0 + r) * DIM + kk + cp;
            uint4 pk;
            pk.x = pack2(wp[0], wp[1]);
            pk.y = pack2(wp[2], wp[3]);
            pk.z = pack2(wp[4], wp[5]);
            pk.w = pack2(wp[6], wp[7]);
            *(uint4*)(&Bs[r][cp]) = pk;
        }
#if HAVE_ASYNC_LDS
        __builtin_amdgcn_s_wait_asynccnt(0);
#endif
        __syncthreads();
        v16bf a0 = load_frag(&As[wm * 32 +  0][0], 40, lane);
        v16bf a1 = load_frag(&As[wm * 32 + 16][0], 40, lane);
        v16bf b0 = load_frag(&Bs[wn * 32 +  0][0], 40, lane);
        v16bf b1 = load_frag(&Bs[wn * 32 + 16][0], 40, lane);
        acc[0][0] = WMMA_BF16(a0, b0, acc[0][0]);
        acc[0][1] = WMMA_BF16(a0, b1, acc[0][1]);
        acc[1][0] = WMMA_BF16(a1, b0, acc[1][0]);
        acc[1][1] = WMMA_BF16(a1, b1, acc[1][1]);
    }

    // Epilogue: per lane the 8 accumulator rows map to 8 consecutive hw positions
    // in one channel plane -> one packed 16B bf16 store / two float4 f32 stores.
    int nl = lane & 15;
    int mB = (lane >> 4) << 3;
#pragma unroll
    for (int i = 0; i < 2; i++)
#pragma unroll
        for (int j = 0; j < 2; j++) {
            int gn  = n0 + wn * 32 + j * 16 + nl;
            float bv = bias[gn];
            int gm0 = m0 + wm * 32 + i * 16 + mB;     // multiple of 8, same batch
            int bb  = gm0 >> 14;
            int hw0 = gm0 & (HWSZ - 1);
            size_t idx = (size_t)bb * DIM * HWSZ + (size_t)gn * HWSZ + hw0;
            if (MODE == 0) {
                uint4 pk;
                pk.x = pack2(acc[i][j][0] + bv, acc[i][j][1] + bv);
                pk.y = pack2(acc[i][j][2] + bv, acc[i][j][3] + bv);
                pk.z = pack2(acc[i][j][4] + bv, acc[i][j][5] + bv);
                pk.w = pack2(acc[i][j][6] + bv, acc[i][j][7] + bv);
                *(uint4*)(outBf + idx) = pk;
            } else {
                float4 r0 = *(const float4*)(resid + idx);
                float4 r1 = *(const float4*)(resid + idx + 4);
                float4 o0, o1;
                o0.x = acc[i][j][0] + bv + r0.x;
                o0.y = acc[i][j][1] + bv + r0.y;
                o0.z = acc[i][j][2] + bv + r0.z;
                o0.w = acc[i][j][3] + bv + r0.w;
                o1.x = acc[i][j][4] + bv + r1.x;
                o1.y = acc[i][j][5] + bv + r1.y;
                o1.z = acc[i][j][6] + bv + r1.z;
                o1.w = acc[i][j][7] + bv + r1.w;
                *(float4*)(outF + idx)     = o0;
                *(float4*)(outF + idx + 4) = o1;
            }
        }
}

// ---------------------------- depthwise 3x3 (SAME) on bf16 NCHW; fused ssq reduction
__global__ void k_dwconv3(const unsigned short* __restrict__ in, const float* __restrict__ wd,
                          const float* __restrict__ bd, unsigned short* __restrict__ out,
                          float* __restrict__ ssq, int doSsq) {
    int tid = threadIdx.x;
    int pix = blockIdx.x * 256 + tid;
    int y = pix >> 7, x = pix & 127;
    int ch = blockIdx.y, bb = blockIdx.z;
    const unsigned short* plane = in + ((size_t)bb * DIM + ch) * HWSZ;
    const float* wp = wd + ch * 9;
    float acc = bd[ch];
#pragma unroll
    for (int dy = -1; dy <= 1; dy++)
#pragma unroll
        for (int dx = -1; dx <= 1; dx++) {
            int yy = y + dy, xx = x + dx;
            if (yy >= 0 && yy < 128 && xx >= 0 && xx < 128)
                acc += bf2f(plane[yy * 128 + xx]) * wp[(dy + 1) * 3 + (dx + 1)];
        }
    out[((size_t)bb * DIM + ch) * HWSZ + pix] = f2bf(acc);
    if (doSsq) {
        float s = acc * acc;
#pragma unroll
        for (int off = 16; off > 0; off >>= 1) s += __shfl_xor(s, off, 32);
        if ((tid & 31) == 0) atomicAdd(&ssq[bb * DIM + ch], s);
    }
}

// ------------- attn[c][d] = (sum_n q*k) * rsqrt(ssq_q[c]) * rsqrt(ssq_k[d]) * T
// One block per (b,head); 8 waves split K=16384; each wave owns all 3x3 WMMA tiles.
__global__ void k_attn_qk(const unsigned short* __restrict__ q, const unsigned short* __restrict__ k,
                          const float* __restrict__ ssq_q, const float* __restrict__ ssq_k,
                          const float* __restrict__ temp, float* __restrict__ attn) {
    __shared__ float red[CH * CH];
    int tid  = threadIdx.x;
    int lane = tid & 31, wave = tid >> 5;
    int bh = blockIdx.x;
    int bb = bh >> 3, hd = bh & 7;
    int ch0 = hd * CH;
    const unsigned short* qb = q + ((size_t)bb * DIM + ch0) * HWSZ;
    const unsigned short* kb = k + ((size_t)bb * DIM + ch0) * HWSZ;
    v8f acc[3][3];
    for (int i = 0; i < 3; i++)
        for (int j = 0; j < 3; j++) acc[i][j] = vzero8();
    for (int n = wave * 32; n < HWSZ; n += 256) {
        if (n + 256 < HWSZ) {                      // stream next K-slice into cache
            __builtin_prefetch(qb + n + 256, 0, 0);
            __builtin_prefetch(kb + n + 256, 0, 0);
        }
        v16bf af[3], bf[3];
#pragma unroll
        for (int i = 0; i < 3; i++) af[i] = load_frag(qb + (size_t)(i * 16) * HWSZ + n, HWSZ, lane);
#pragma unroll
        for (int j = 0; j < 3; j++) bf[j] = load_frag(kb + (size_t)(j * 16) * HWSZ + n, HWSZ, lane);
#pragma unroll
        for (int i = 0; i < 3; i++)
#pragma unroll
            for (int j = 0; j < 3; j++)
                acc[i][j] = WMMA_BF16(af[i], bf[j], acc[i][j]);
    }
    for (int i = tid; i < CH * CH; i += 256) red[i] = 0.f;
    __syncthreads();
    int nl = lane & 15, mB = (lane >> 4) << 3;
#pragma unroll
    for (int i = 0; i < 3; i++)
#pragma unroll
        for (int j = 0; j < 3; j++)
#pragma unroll
            for (int r = 0; r < 8; r++) {
                int c = i * 16 + mB + r;
                int d = j * 16 + nl;
                atomicAdd(&red[c * CH + d], acc[i][j][r]);   // ds_add_f32
            }
    __syncthreads();
    float T = temp[0];
    for (int idx = tid; idx < CH * CH; idx += 256) {
        int c = idx / CH, d = idx % CH;
        float sc = rsqrtf(fmaxf(ssq_q[bb * DIM + ch0 + c], 1e-24f));
        float sd = rsqrtf(fmaxf(ssq_k[bb * DIM + ch0 + d], 1e-24f));
        attn[(size_t)bh * CH * CH + idx] = red[idx] * sc * sd * T;
    }
}

// ----- per-row top-k thresholds (k = 24,32,36,38) -> 4 masked softmaxes folded into one matrix
__global__ void k_topk_combine(const float* __restrict__ attn, const float* __restrict__ a1,
                               const float* __restrict__ a2, const float* __restrict__ a3,
                               const float* __restrict__ a4, float* __restrict__ acomb) {
    int row = threadIdx.x;
    if (row >= CH) return;
    int bh = blockIdx.x;
    const float* src = attn + (size_t)bh * CH * CH + row * CH;
    float v[CH], s[CH];
    for (int i = 0; i < CH; i++) { v[i] = src[i]; s[i] = v[i]; }
    for (int i = 1; i < CH; i++) {                 // insertion sort, descending
        float key = s[i]; int j = i - 1;
        while (j >= 0 && s[j] < key) { s[j + 1] = s[j]; j--; }
        s[j + 1] = key;
    }
    const int kks[4] = {24, 32, 36, 38};           // C/2, 2C/3, 3C/4, 4C/5
    float aw[4] = {a1[0], a2[0], a3[0], a4[0]};
    float m = s[0];
    float th[4], inv[4];
    for (int t = 0; t < 4; t++) {
        th[t] = s[kks[t] - 1];
        float sum = 0.f;
        for (int d = 0; d < CH; d++)
            if (v[d] >= th[t]) sum += __expf(v[d] - m);
        inv[t] = 1.f / sum;
    }
    float* dst = acomb + (size_t)bh * CH * CH + row * CH;
    for (int d = 0; d < CH; d++) {
        float e = __expf(v[d] - m);
        float o = 0.f;
        for (int t = 0; t < 4; t++)
            if (v[d] >= th[t]) o += aw[t] * e * inv[t];
        dst[d] = o;
    }
}

// ---------------- out = A_comb(48x48) @ v(48x16384) per (b,h); K padded 48->64
__global__ void k_attn_v(const float* __restrict__ acomb, const unsigned short* __restrict__ v,
                         unsigned short* __restrict__ fusion) {
    __shared__ unsigned short As[CH][72];    // [c][k], zero-padded k 48..63
    __shared__ unsigned short Vs[256][72];   // transposed: [n][k]
    int tid  = threadIdx.x;
    int lane = tid & 31, wave = tid >> 5;
    int bh = blockIdx.y;
    int bb = bh >> 3, hd = bh & 7;
    int ch0 = hd * CH;
    int nb  = blockIdx.x * 256;
    const float* am = acomb + (size_t)bh * CH * CH;
    for (int idx = tid; idx < CH * 64; idx += 256) {
        int c = idx >> 6, d = idx & 63;
        As[c][d] = (d < CH) ? f2bf(am[c * CH + d]) : (unsigned short)0;
    }
    const unsigned short* vb = v + ((size_t)bb * DIM + ch0) * HWSZ + nb;
    for (int d = 0; d < CH; d++) Vs[tid][d] = vb[(size_t)d * HWSZ + tid];
#pragma unroll
    for (int d = CH; d < 64; d++) Vs[tid][d] = 0;
    __syncthreads();

    v8f acc[3][2];
    for (int i = 0; i < 3; i++)
        for (int j = 0; j < 2; j++) acc[i][j] = vzero8();
#pragma unroll
    for (int ks = 0; ks < 2; ks++) {
        int kOff = ks * 32;
        v16bf af[3], bf[2];
#pragma unroll
        for (int i = 0; i < 3; i++) af[i] = load_frag(&As[i * 16][kOff], 72, lane);
#pragma unroll
        for (int j = 0; j < 2; j++) bf[j] = load_frag(&Vs[wave * 32 + j * 16][kOff], 72, lane);
#pragma unroll
        for (int i = 0; i < 3; i++)
#pragma unroll
            for (int j = 0; j < 2; j++)
                acc[i][j] = WMMA_BF16(af[i], bf[j], acc[i][j]);
    }
    // per lane: 8 consecutive channels at one pixel -> single packed 16B store
    int nl = lane & 15, mB = (lane >> 4) << 3;
#pragma unroll
    for (int i = 0; i < 3; i++)
#pragma unroll
        for (int j = 0; j < 2; j++) {
            int col = nb + wave * 32 + j * 16 + nl;
            size_t p = (size_t)bb * HWSZ + col;
            uint4 pk;
            pk.x = pack2(acc[i][j][0], acc[i][j][1]);
            pk.y = pack2(acc[i][j][2], acc[i][j][3]);
            pk.z = pack2(acc[i][j][4], acc[i][j][5]);
            pk.w = pack2(acc[i][j][6], acc[i][j][7]);
            *(uint4*)(fusion + p * DIM + ch0 + i * 16 + mB) = pk;
        }
}

// =======================================================================================
extern "C" void kernel_launch(void* const* d_in, const int* in_sizes, int n_in,
                              void* d_out, int out_size, void* d_ws, size_t ws_size,
                              hipStream_t stream) {
    const float* T_E   = (const float*)d_in[0];
    const float* C_E   = (const float*)d_in[1];
    const float* tn_w  = (const float*)d_in[2];
    const float* tn_b  = (const float*)d_in[3];
    const float* cn_w  = (const float*)d_in[4];
    const float* cn_b  = (const float*)d_in[5];
    const float* tq_w  = (const float*)d_in[6];
    const float* tq_b  = (const float*)d_in[7];
    const float* tq_dw = (const float*)d_in[8];
    const float* tq_db = (const float*)d_in[9];
    const float* ck_w  = (const float*)d_in[10];
    const float* ck_b  = (const float*)d_in[11];
    const float* ck_dw = (const float*)d_in[12];
    const float* ck_db = (const float*)d_in[13];
    const float* cv_w  = (const float*)d_in[14];
    const float* cv_b  = (const float*)d_in[15];
    const float* cv_dw = (const float*)d_in[16];
    const float* cv_db = (const float*)d_in[17];
    const float* temp  = (const float*)d_in[18];
    const float* pj_w  = (const float*)d_in[19];
    const float* pj_b  = (const float*)d_in[20];
    const float* a1    = (const float*)d_in[21];
    const float* a2    = (const float*)d_in[22];
    const float* a3    = (const float*)d_in[23];
    const float* a4    = (const float*)d_in[24];

    const size_t BUF = (size_t)MTOT * DIM * sizeof(unsigned short);  // 100.7 MB
    char* w = (char*)d_ws;
    unsigned short* buf0 = (unsigned short*)w; w += BUF;  // tn_act   -> q_dw
    unsigned short* buf1 = (unsigned short*)w; w += BUF;  // cn_act   -> k_dw
    unsigned short* buf2 = (unsigned short*)w; w += BUF;  // g_q      -> fusion
    unsigned short* buf3 = (unsigned short*)w; w += BUF;  // g_k      -> v_dw
    unsigned short* buf4 = (unsigned short*)w; w += BUF;  // g_v
    float* ssq   = (float*)w; w += 2 * BATCH * DIM * sizeof(float);
    float* ssq_q = ssq;
    float* ssq_k = ssq + BATCH * DIM;
    float* attn  = (float*)w; w += (size_t)BATCH * HEADS * CH * CH * sizeof(float);
    float* acomb = (float*)w; w += (size_t)BATCH * HEADS * CH * CH * sizeof(float);

    k_zero<<<(2 * BATCH * DIM + 255) / 256, 256, 0, stream>>>(ssq, 2 * BATCH * DIM);

    k_layernorm<<<MTOT / 8, 256, 0, stream>>>(T_E, tn_w, tn_b, buf0);
    k_layernorm<<<MTOT / 8, 256, 0, stream>>>(C_E, cn_w, cn_b, buf1);

    dim3 gg(MTOT / 128, DIM / 64);
    k_gemm384<0><<<gg, 256, 0, stream>>>(buf0, tq_w, tq_b, buf2, nullptr, nullptr);
    k_gemm384<0><<<gg, 256, 0, stream>>>(buf1, ck_w, ck_b, buf3, nullptr, nullptr);
    k_gemm384<0><<<gg, 256, 0, stream>>>(buf1, cv_w, cv_b, buf4, nullptr, nullptr);

    dim3 gd(HWSZ / 256, DIM, BATCH);
    k_dwconv3<<<gd, 256, 0, stream>>>(buf2, tq_dw, tq_db, buf0, ssq_q, 1);
    k_dwconv3<<<gd, 256, 0, stream>>>(buf3, ck_dw, ck_db, buf1, ssq_k, 1);
    k_dwconv3<<<gd, 256, 0, stream>>>(buf4, cv_dw, cv_db, buf3, nullptr, 0);

    k_attn_qk<<<BATCH * HEADS, 256, 0, stream>>>(buf0, buf1, ssq_q, ssq_k, temp, attn);
    k_topk_combine<<<BATCH * HEADS, 64, 0, stream>>>(attn, a1, a2, a3, a4, acomb);

    dim3 gv(HWSZ / 256, BATCH * HEADS);
    k_attn_v<<<gv, 256, 0, stream>>>(acomb, buf3, buf2);

    k_gemm384<1><<<gg, 256, 0, stream>>>(buf2, pj_w, pj_b, nullptr, (float*)d_out, C_E);
}
